// DecoderOnlyBlock_38139309588692
// MI455X (gfx1250) — compile-verified, matrix operand order checked by reading, not run on hardware
//
#include <hip/hip_runtime.h>
#include <hip/hip_bf16.h>
#include <math.h>

typedef _Float16 v8h  __attribute__((ext_vector_type(8)));
typedef _Float16 v16h __attribute__((ext_vector_type(16)));
typedef float    v8f  __attribute__((ext_vector_type(8)));
typedef int      v4i  __attribute__((ext_vector_type(4)));

#define HH   8
#define HD   96
#define SEQ  4096
#define DM   768
#define DFF  3072
#define NBLK 64
#define MKB  7
#define ROWS 16384   // b*S

#if defined(__has_builtin)
#  if __has_builtin(__builtin_amdgcn_global_load_async_to_lds_b128)
#    define HAS_ASYNC_LDS 1
#  endif
#  if __has_builtin(__builtin_amdgcn_s_wait_asynccnt)
#    define HAS_WAIT_ASYNC 1
#  endif
#endif

// 16-byte global -> LDS copy. Uses CDNA5 async DMA path (ASYNCcnt) when the
// toolchain exposes it; falls back to a VGPR round-trip copy otherwise.
__device__ __forceinline__ void async_copy16(const _Float16* g, _Float16* l) {
#if defined(HAS_ASYNC_LDS)
    __builtin_amdgcn_global_load_async_to_lds_b128(
        (__attribute__((address_space(1))) v4i*)(g),
        (__attribute__((address_space(3))) v4i*)(l), 0, 0);
#else
    *(v8h*)l = *(const v8h*)g;
#endif
}

__device__ __forceinline__ void async_wait0() {
#if defined(HAS_ASYNC_LDS)
#  if defined(HAS_WAIT_ASYNC)
    __builtin_amdgcn_s_wait_asynccnt(0);
#  else
    asm volatile("s_wait_asynccnt 0x0" ::: "memory");
#  endif
#endif
}

__device__ __forceinline__ void async_wait4() {
#if defined(HAS_ASYNC_LDS)
#  if defined(HAS_WAIT_ASYNC)
    __builtin_amdgcn_s_wait_asynccnt(4);
#  else
    asm volatile("s_wait_asynccnt 0x4" ::: "memory");
#  endif
#endif
}

// Build a 16x32 f16 WMMA operand fragment: lane holds 8 halfs at K=kb..kb+7
// and 8 halfs at K=kb+16..kb+23, per CDNA5 ISA operand layout.
__device__ __forceinline__ v16h make_frag(const _Float16* p) {
    v8h lo = *(const v8h*)p;
    v8h hi = *(const v8h*)(p + 16);
    v16h f;
#pragma unroll
    for (int i = 0; i < 8; ++i) { f[i] = lo[i]; f[i + 8] = hi[i]; }
    return f;
}

__device__ __forceinline__ v8f wmma32(v16h a, v16h b, v8f c) {
    return __builtin_amdgcn_wmma_f32_16x16x32_f16(false, a, false, b, (short)0, c,
                                                  false, false);
}

// ---------------- weight cast + transpose: out[n*K + k] = (f16) W[k*N + n]
__global__ __launch_bounds__(256) void castT_kernel(const float* __restrict__ W,
                                                    _Float16* __restrict__ out,
                                                    int K, int N) {
    int idx = blockIdx.x * 256 + threadIdx.x;
    if (idx < K * N) {
        int n = idx / K, k = idx % K;
        out[idx] = (_Float16)W[(size_t)k * N + n];
    }
}

// ---------------- LayerNorm (row of 768) -> f16
__global__ __launch_bounds__(256) void ln_kernel(const float* __restrict__ x,
                                                 const float* __restrict__ g,
                                                 const float* __restrict__ bta,
                                                 _Float16* __restrict__ out) {
    __shared__ float red[256];
    const int row = blockIdx.x, tid = threadIdx.x;
    const float* xr = x + (size_t)row * DM;
    float vals[3], s = 0.f;
#pragma unroll
    for (int i = 0; i < 3; ++i) { vals[i] = xr[tid + i * 256]; s += vals[i]; }
    red[tid] = s; __syncthreads();
    for (int off = 128; off > 0; off >>= 1) {
        if (tid < off) red[tid] += red[tid + off];
        __syncthreads();
    }
    float mu = red[0] * (1.0f / DM);
    __syncthreads();
    float v2 = 0.f;
#pragma unroll
    for (int i = 0; i < 3; ++i) { float d = vals[i] - mu; v2 += d * d; }
    red[tid] = v2; __syncthreads();
    for (int off = 128; off > 0; off >>= 1) {
        if (tid < off) red[tid] += red[tid + off];
        __syncthreads();
    }
    float rstd = rsqrtf(red[0] * (1.0f / DM) + 1e-5f);
#pragma unroll
    for (int i = 0; i < 3; ++i) {
        int c = tid + i * 256;
        out[(size_t)row * DM + c] = (_Float16)((vals[i] - mu) * rstd * g[c] + bta[c]);
    }
}

// Stage one 128x32 A tile + 128x32 B tile (4 async b128 ops per thread/wave).
__device__ __forceinline__ void copy_tile(const _Float16* gA, const _Float16* gB,
                                          _Float16* lA, _Float16* lB, int tid,
                                          int K) {
#pragma unroll
    for (int c = tid; c < 512; c += 256) {
        int row = c >> 2, off = (c & 3) * 8;
        async_copy16(gA + (size_t)row * K + off, lA + row * 32 + off);
        async_copy16(gB + (size_t)row * K + off, lB + row * 32 + off);
    }
}

// ---------------- generic 128x128 WMMA GEMM, A[M][K] f16, Bt[N][K] f16
// Double-buffered LDS with one-tile-ahead async global->LDS prefetch.
// mode 0: QKV scatter  mode 1: gelu->f16  mode 2: residual+bias->f32
__global__ __launch_bounds__(256) void gemm_kernel(
    const _Float16* __restrict__ A, const _Float16* __restrict__ Bt,
    int M, int N, int K, int mode,
    const float* __restrict__ bias0, const float* __restrict__ bias1,
    const float* __restrict__ bias2, const float* __restrict__ residual,
    _Float16* __restrict__ o16a, _Float16* __restrict__ o16b,
    _Float16* __restrict__ o16c, float* __restrict__ o32) {
    __shared__ __align__(16) _Float16 As[2][128 * 32];
    __shared__ __align__(16) _Float16 Bs[2][128 * 32];

    const int tid  = threadIdx.x;
    const int lane = tid & 31;
    const int wave = tid >> 5;
    const int tileM0 = blockIdx.x * 128;
    const int tileN0 = blockIdx.y * 128;
    const int wm = (wave & 3) * 32;
    const int wn = (wave >> 2) * 64;
    const int r   = lane & 15;
    const int kb8 = (lane >> 4) * 8;

    const _Float16* gA = A + (size_t)tileM0 * K;
    const _Float16* gB = Bt + (size_t)tileN0 * K;

    v8f acc[2][4];
#pragma unroll
    for (int i = 0; i < 2; ++i)
#pragma unroll
        for (int j = 0; j < 4; ++j)
#pragma unroll
            for (int v = 0; v < 8; ++v) acc[i][j][v] = 0.f;

    copy_tile(gA, gB, As[0], Bs[0], tid, K);   // prologue: tile 0 in flight

    const int NK = K >> 5;
    for (int kt = 0; kt < NK; ++kt) {
        const int cur = kt & 1;
        if (kt + 1 < NK) {
            copy_tile(gA + (kt + 1) * 32, gB + (kt + 1) * 32,
                      As[cur ^ 1], Bs[cur ^ 1], tid, K);
            async_wait4();   // in-order: <=4 outstanding => current tile landed
        } else {
            async_wait0();
        }
        __syncthreads();

        v16h af[2], bf[4];
#pragma unroll
        for (int i = 0; i < 2; ++i)
            af[i] = make_frag(&As[cur][(wm + i * 16 + r) * 32 + kb8]);
#pragma unroll
        for (int j = 0; j < 4; ++j)
            bf[j] = make_frag(&Bs[cur][(wn + j * 16 + r) * 32 + kb8]);
#pragma unroll
        for (int i = 0; i < 2; ++i)
#pragma unroll
            for (int j = 0; j < 4; ++j) acc[i][j] = wmma32(af[i], bf[j], acc[i][j]);
        __syncthreads();   // all waves done reading before this buf is re-filled
    }

    // epilogue: D element v -> row = tile + v + 8*(lane>>4), col = tile + (lane&15)
    const int rhalf = (lane >> 4) * 8;
    const int cN = lane & 15;
#pragma unroll
    for (int i = 0; i < 2; ++i)
#pragma unroll
        for (int j = 0; j < 4; ++j) {
            int mbase = tileM0 + wm + i * 16 + rhalf;
            int col   = tileN0 + wn + j * 16 + cN;
#pragma unroll
            for (int v = 0; v < 8; ++v) {
                int row = mbase + v;
                float val = acc[i][j][v];
                if (mode == 0) {
                    int which = col / DM, nn = col % DM;
                    int h = nn / HD, dh = nn % HD;
                    int bb = row >> 12, s = row & (SEQ - 1);
                    const float* bp = which == 0 ? bias0 : (which == 1 ? bias1 : bias2);
                    _Float16 hv = (_Float16)(val + bp[nn]);
                    size_t bh = (size_t)(bb * HH + h);
                    if (which == 0)       o16a[(bh * SEQ + s) * HD + dh] = hv;
                    else if (which == 1)  o16b[(bh * SEQ + s) * HD + dh] = hv;
                    else                  o16c[(bh * HD + dh) * SEQ + s] = hv;  // V^T
                } else if (mode == 1) {
                    float t = val + bias0[col];
                    float gel = 0.5f * t *
                        (1.0f + tanhf(0.7978845608f * (t + 0.044715f * t * t * t)));
                    o16a[(size_t)row * N + col] = (_Float16)gel;
                } else {
                    o32[(size_t)row * N + col] =
                        residual[(size_t)row * N + col] + val + bias0[col];
                }
            }
        }
}

// ---------------- BigBird attention: one workgroup per (qblock, head, batch)
__global__ __launch_bounds__(256) void attn_kernel(
    const _Float16* __restrict__ Qf, const _Float16* __restrict__ Kf,
    const _Float16* __restrict__ Vt, const int* __restrict__ kbi,
    _Float16* __restrict__ attn_out) {
    __shared__ __align__(16) _Float16 Qs[64 * HD];
    __shared__ __align__(16) _Float16 Ks[64 * HD];
    __shared__ __align__(16) float    Ss[64 * 64];
    __shared__ __align__(16) _Float16 Ps[64 * 64];
    __shared__ float rowmax[64], rowsum[64], rowscale[64];
    __shared__ float red[64][4];

    const int qb = blockIdx.x, h = blockIdx.y, bb = blockIdx.z;
    const int tid = threadIdx.x, lane = tid & 31, wave = tid >> 5;
    const int r   = lane & 15;
    const int kb8 = (lane >> 4) * 8;
    const size_t bh = (size_t)(bb * HH + h);

    const _Float16* Qbase = Qf + (bh * SEQ + (size_t)qb * 64) * HD;
#pragma unroll
    for (int c = tid; c < 768; c += 256)
        async_copy16(&Qbase[c * 8], &Qs[c * 8]);
    if (tid < 64) { rowmax[tid] = -3.0e38f; rowsum[tid] = 0.f; }
    __syncthreads();

    const int mt  = (wave & 3) * 16;        // wave M tile (rows)
    const int ntb = (wave >> 2) * 48;       // wave O column base (3 tiles of 16)
    v8f oacc[3];
#pragma unroll
    for (int t = 0; t < 3; ++t)
#pragma unroll
        for (int v = 0; v < 8; ++v) oacc[t][v] = 0.f;

    for (int j = 0; j < MKB; ++j) {
        int kblk = kbi[qb * MKB + j];
        if (kblk < 0) continue;  // uniform across block

        const _Float16* Kbase = Kf + (bh * SEQ + (size_t)kblk * 64) * HD;
#pragma unroll
        for (int c = tid; c < 768; c += 256)
            async_copy16(&Kbase[c * 8], &Ks[c * 8]);
        async_wait0();           // Q (first pass) + K block landed in LDS
        __syncthreads();

        // ---- scores S = Q K^T / sqrt(hd): 16 tiles, 2 per wave
        v8f sacc[2];
#pragma unroll
        for (int t = 0; t < 2; ++t)
#pragma unroll
            for (int v = 0; v < 8; ++v) sacc[t][v] = 0.f;
        for (int k0 = 0; k0 < HD; k0 += 32) {
            v16h aq = make_frag(&Qs[(mt + r) * HD + k0 + kb8]);
#pragma unroll
            for (int t = 0; t < 2; ++t) {
                int nt = ((wave >> 2) * 2 + t) * 16;
                v16h bk = make_frag(&Ks[(nt + r) * HD + k0 + kb8]);
                sacc[t] = wmma32(aq, bk, sacc[t]);
            }
        }
        const float sc = 0.1020620726159658f;  // 1/sqrt(96)
#pragma unroll
        for (int t = 0; t < 2; ++t) {
            int colg = ((wave >> 2) * 2 + t) * 16 + (lane & 15);
            int rowb = mt + (lane >> 4) * 8;
#pragma unroll
            for (int v = 0; v < 8; ++v) {
                int rowg = rowb + v;
                float val = sacc[t][v] * sc;
                if (kblk == qb && colg > rowg) val = -1.0e9f;  // causal (diag block)
                Ss[rowg * 64 + colg] = val;
            }
        }
        __syncthreads();

        // ---- online softmax update (4 threads per row, 16 cols each)
        const int srow = tid >> 2, seg = tid & 3;
        float sv[16], lm = -3.0e38f;
#pragma unroll
        for (int c = 0; c < 16; ++c) {
            sv[c] = Ss[srow * 64 + seg * 16 + c];
            lm = fmaxf(lm, sv[c]);
        }
        red[srow][seg] = lm;
        __syncthreads();
        float mprev = rowmax[srow];
        float mnew = fmaxf(fmaxf(red[srow][0], red[srow][1]),
                           fmaxf(red[srow][2], red[srow][3]));
        mnew = fmaxf(mnew, mprev);
        float ls = 0.f;
#pragma unroll
        for (int c = 0; c < 16; ++c) {
            float pv = __expf(sv[c] - mnew);
            ls += pv;
            Ps[srow * 64 + seg * 16 + c] = (_Float16)pv;
        }
        __syncthreads();
        red[srow][seg] = ls;
        __syncthreads();
        if (seg == 0) {
            float sum4 = red[srow][0] + red[srow][1] + red[srow][2] + red[srow][3];
            float scale = __expf(mprev - mnew);
            rowscale[srow] = scale;
            rowsum[srow] = rowsum[srow] * scale + sum4;
            rowmax[srow] = mnew;
        }
        __syncthreads();

        // ---- O = O*scale + P @ V   (B operand from transposed V in global)
        {
            int rowb = mt + (lane >> 4) * 8;
#pragma unroll
            for (int t = 0; t < 3; ++t)
#pragma unroll
                for (int v = 0; v < 8; ++v) oacc[t][v] *= rowscale[rowb + v];
        }
        const _Float16* Vbase = Vt + bh * HD * SEQ + (size_t)kblk * 64;
        for (int k0 = 0; k0 < 64; k0 += 32) {
            v16h ap = make_frag(&Ps[(mt + r) * 64 + k0 + kb8]);
#pragma unroll
            for (int t = 0; t < 3; ++t) {
                int d = ntb + t * 16 + r;
                v16h bv = make_frag(Vbase + (size_t)d * SEQ + k0 + kb8);
                oacc[t] = wmma32(ap, bv, oacc[t]);
            }
        }
        __syncthreads();  // protect Ks/Ss/Ps for next key block
    }

    // ---- normalize and store f16 [b][s][h*96+d]
#pragma unroll
    for (int t = 0; t < 3; ++t) {
        int d = ntb + t * 16 + (lane & 15);
        int rowb = mt + (lane >> 4) * 8;
#pragma unroll
        for (int v = 0; v < 8; ++v) {
            int q = rowb + v;
            float o = oacc[t][v] / rowsum[q];
            attn_out[((size_t)bb * SEQ + (size_t)qb * 64 + q) * DM + h * HD + d] =
                (_Float16)o;
        }
    }
}

extern "C" void kernel_launch(void* const* d_in, const int* in_sizes, int n_in,
                              void* d_out, int out_size, void* d_ws, size_t ws_size,
                              hipStream_t stream) {
    const float* qin  = (const float*)d_in[0];
    const float* ln1g = (const float*)d_in[1];
    const float* ln1b = (const float*)d_in[2];
    const float* Wq   = (const float*)d_in[3];
    const float* bq   = (const float*)d_in[4];
    const float* Wk   = (const float*)d_in[5];
    const float* bk   = (const float*)d_in[6];
    const float* Wv   = (const float*)d_in[7];
    const float* bv   = (const float*)d_in[8];
    const float* Wo   = (const float*)d_in[9];
    const float* bo   = (const float*)d_in[10];
    const float* ln2g = (const float*)d_in[11];
    const float* ln2b = (const float*)d_in[12];
    const float* W1   = (const float*)d_in[13];
    const float* b1   = (const float*)d_in[14];
    const float* W2   = (const float*)d_in[15];
    const float* b2   = (const float*)d_in[16];
    const int*   kbi  = (const int*)d_in[17];

    const size_t ACT = (size_t)ROWS * DM;           // 12,582,912 halfs
    _Float16* ws16  = (_Float16*)d_ws;
    _Float16* WqkvT = ws16;                          // [2304][768]
    _Float16* WoT   = WqkvT + (size_t)3 * DM * DM;   // [768][768]
    _Float16* W1T   = WoT + (size_t)DM * DM;         // [3072][768]
    _Float16* W2T   = W1T + (size_t)DFF * DM;        // [768][3072]
    _Float16* h1    = W2T + (size_t)DM * DFF;        // LN out (reused for LN2)
    _Float16* Qf    = h1 + ACT;                      // [b][h][s][d]
    _Float16* Kf    = Qf + ACT;
    _Float16* Vt    = Kf + ACT;                      // [b][h][d][s]
    _Float16* attnb = Vt + ACT;                      // [b*s][768]
    _Float16* mid   = Qf;                            // overlay: 16384*3072 == 4*ACT
    float*    xbuf  = (float*)(attnb + ACT);

    const int E1 = DM * DM, E2 = DM * DFF;
    castT_kernel<<<(E1 + 255) / 256, 256, 0, stream>>>(Wq, WqkvT, DM, DM);
    castT_kernel<<<(E1 + 255) / 256, 256, 0, stream>>>(Wk, WqkvT + (size_t)DM * DM, DM, DM);
    castT_kernel<<<(E1 + 255) / 256, 256, 0, stream>>>(Wv, WqkvT + (size_t)2 * DM * DM, DM, DM);
    castT_kernel<<<(E1 + 255) / 256, 256, 0, stream>>>(Wo, WoT, DM, DM);
    castT_kernel<<<(E2 + 255) / 256, 256, 0, stream>>>(W1, W1T, DM, DFF);
    castT_kernel<<<(E2 + 255) / 256, 256, 0, stream>>>(W2, W2T, DFF, DM);

    // x1 = LN1(q) -> f16
    ln_kernel<<<ROWS, 256, 0, stream>>>(qin, ln1g, ln1b, h1);
    // QKV fused GEMM with scatter epilogue (V transposed)
    gemm_kernel<<<dim3(ROWS / 128, 2304 / 128), 256, 0, stream>>>(
        h1, WqkvT, ROWS, 2304, DM, 0, bq, bk, bv, nullptr, Qf, Kf, Vt, nullptr);
    // BigBird attention
    attn_kernel<<<dim3(NBLK, HH, 4), 256, 0, stream>>>(Qf, Kf, Vt, kbi, attnb);
    // x = q + attn @ Wo + bo
    gemm_kernel<<<dim3(ROWS / 128, DM / 128), 256, 0, stream>>>(
        attnb, WoT, ROWS, DM, DM, 2, bo, nullptr, nullptr, qin,
        nullptr, nullptr, nullptr, xbuf);
    // h = LN2(x)
    ln_kernel<<<ROWS, 256, 0, stream>>>(xbuf, ln2g, ln2b, h1);
    // mid = gelu(h @ W1 + b1)
    gemm_kernel<<<dim3(ROWS / 128, DFF / 128), 256, 0, stream>>>(
        h1, W1T, ROWS, DFF, DM, 1, b1, nullptr, nullptr, nullptr,
        mid, nullptr, nullptr, nullptr);
    // out = x + mid @ W2 + b2
    gemm_kernel<<<dim3(ROWS / 128, DM / 128), 256, 0, stream>>>(
        mid, W2T, ROWS, DM, DFF, 2, b2, nullptr, nullptr, xbuf,
        nullptr, nullptr, nullptr, (float*)d_out);
    (void)in_sizes; (void)n_in; (void)out_size; (void)ws_size;
}
